// DGCNN_79517024518279
// MI455X (gfx1250) — compile-verified, matrix operand order checked by reading, not run on hardware
//
#include <hip/hip_runtime.h>
#include <hip/hip_bf16.h>
#include <math.h>

#define NPTS  16384
#define BATCH 8
#define PPTS  2048
#define KNN   20

typedef __attribute__((ext_vector_type(2)))  float        v2f;
typedef __attribute__((ext_vector_type(8)))  float        v8f;
typedef __attribute__((ext_vector_type(16))) _Float16     v16h;
typedef __attribute__((ext_vector_type(2)))  _Float16     h2v;
typedef __attribute__((ext_vector_type(4)))  unsigned int u32x4;

union AFrag {
  v16h v;
  h2v h[8];
  unsigned int u[8];
  u32x4 q[2];
};

// K position inside a 32-wide chunk for 16-bit A/B fragment register v,
// lane-half p (0 = lanes 0-15, 1 = lanes 16-31). Per ISA 7.12.2:
// VGPR0-3: K=0..7 (p=0) / 8..15 (p=1); VGPR4-7: K=16..23 / 24..31.
__device__ __forceinline__ int kposf(int v, int p) {
  return 2 * (v & 3) + 8 * p + 16 * (v >> 2);
}

// ---------------------------------------------------------------------------
// Pack a row-major f32 weight matrix W[Kreal][Nreal] into the CDNA5 16-bit
// B-fragment layout, lane-major so consumers load 2x b128 per fragment:
//   out[((ch*Ntiles + nt)*32 + lane)*8 + v] = f16x2(W[K], W[K+1]) for col(lane).
// ---------------------------------------------------------------------------
__global__ __launch_bounds__(256)
void pack_b_kernel(const float* __restrict__ W, int Kreal, int Nreal,
                   int Kch, int Ntiles, unsigned int* __restrict__ out) {
  int t = blockIdx.x * blockDim.x + threadIdx.x;
  int total = Kch * Ntiles * 256;
  if (t >= total) return;
  int lane = t & 31;
  int v    = (t >> 5) & 7;
  int blk  = t >> 8;              // ch*Ntiles + nt
  int nt   = blk % Ntiles;
  int ch   = blk / Ntiles;
  int p = lane >> 4;
  int n = lane & 15;
  int col = nt * 16 + n;
  int k = ch * 32 + kposf(v, p);
  _Float16 lo = ((k     < Kreal) && (col < Nreal)) ? (_Float16)W[(size_t)k * Nreal + col]       : (_Float16)0.f;
  _Float16 hi = ((k + 1 < Kreal) && (col < Nreal)) ? (_Float16)W[(size_t)(k + 1) * Nreal + col] : (_Float16)0.f;
  union { _Float16 h[2]; unsigned int u; } pk;
  pk.h[0] = lo; pk.h[1] = hi;
  out[((size_t)blk * 32 + lane) * 8 + v] = pk.u;
}

// ---------------------------------------------------------------------------
// Build x0 = concat(feats, pos) padded to 8 dims (f32 + f16) and point norms.
// ---------------------------------------------------------------------------
__global__ __launch_bounds__(256)
void prep_x0_kernel(const float* __restrict__ feats, const float* __restrict__ pos,
                    float* __restrict__ x0f, _Float16* __restrict__ x0h,
                    float* __restrict__ norms) {
  int i = blockIdx.x * blockDim.x + threadIdx.x;
  if (i >= NPTS) return;
  float vbuf[8];
  vbuf[0] = feats[i * 3 + 0]; vbuf[1] = feats[i * 3 + 1]; vbuf[2] = feats[i * 3 + 2];
  vbuf[3] = pos[i * 3 + 0];   vbuf[4] = pos[i * 3 + 1];   vbuf[5] = pos[i * 3 + 2];
  vbuf[6] = 0.f; vbuf[7] = 0.f;
  float s = 0.f;
  #pragma unroll
  for (int k = 0; k < 6; ++k) s += vbuf[k] * vbuf[k];
  #pragma unroll
  for (int k = 0; k < 8; ++k) {
    x0f[(size_t)i * 8 + k] = vbuf[k];
    x0h[(size_t)i * 8 + k] = (_Float16)vbuf[k];
  }
  norms[i] = s;
}

// ---------------------------------------------------------------------------
// kNN: one wave per 16-row tile. Gram via v_wmma_f32_16x16x4_f32, top-20 in
// registers per row. Insertion is branch-free (cndmask chain) behind a single
// guard so no divergent array-copy paths are generated.
// ---------------------------------------------------------------------------
template<int D>
__global__ __launch_bounds__(128)
void knn_kernel(const float* __restrict__ X, const float* __restrict__ norms,
                int* __restrict__ idxOut) {
  __shared__ float tl[4][16 * 17];
  const int w    = threadIdx.x >> 5;
  const int lane = threadIdx.x & 31;
  const int wid  = blockIdx.x * 4 + w;
  const int b    = wid >> 7;       // 128 row-tiles per batch
  const int rt   = wid & 127;
  const int rowBase = b * PPTS + rt * 16;
  const int p = lane >> 4;
  const int n = lane & 15;
  float* tlw = tl[w];

  // Preload A fragments for all D/4 K-chunks (f32 16x4: VGPR0=K0/K2, VGPR1=K1/K3)
  float areg[D / 2];
  {
    const float* xa = X + (size_t)(rowBase + n) * D;
    #pragma unroll
    for (int c = 0; c < D / 4; ++c) {
      areg[2 * c]     = xa[4 * c + 2 * p];
      areg[2 * c + 1] = xa[4 * c + 2 * p + 1];
    }
  }
  float xi2[8];
  {
    const float* nb = norms + rowBase + 8 * p;
    #pragma unroll
    for (int v = 0; v < 8; ++v) xi2[v] = nb[v];
  }

  float dk[KNN]; int ik[KNN];
  #pragma unroll
  for (int q = 0; q < KNN; ++q) { dk[q] = 3.4e38f; ik[q] = 0; }

  for (int ct = 0; ct < PPTS / 16; ++ct) {
    const int colBase = b * PPTS + ct * 16;
    const float* xb = X + (size_t)(colBase + n) * D;
    v8f acc = {};
    #pragma unroll
    for (int c = 0; c < D / 4; ++c) {
      v2f av, bv;
      av[0] = areg[2 * c]; av[1] = areg[2 * c + 1];
      bv[0] = xb[4 * c + 2 * p]; bv[1] = xb[4 * c + 2 * p + 1];
      acc = __builtin_amdgcn_wmma_f32_16x16x4_f32(false, av, false, bv,
                                                  (short)0, acc, false, false);
    }
    float xj2 = norms[colBase + n];
    #pragma unroll
    for (int v = 0; v < 8; ++v)
      tlw[(8 * p + v) * 17 + n] = xi2[v] + xj2 - 2.0f * acc[v];
    __syncthreads();
    if (lane < 16) {
      for (int j = 0; j < 16; ++j) {
        float vj = tlw[lane * 17 + j];
        int   ij = colBase + j;
        if (vj < dk[KNN - 1]) {
          // Branch-free sorted insertion: per step, 1 cmp + 4 cndmask.
          #pragma unroll
          for (int q = 0; q < KNN; ++q) {
            const bool  lt = vj < dk[q];
            const float nd = lt ? vj : dk[q];
            const int   ni = lt ? ij : ik[q];
            const float cd = lt ? dk[q] : vj;
            const int   ci = lt ? ik[q] : ij;
            dk[q] = nd; ik[q] = ni;
            vj = cd; ij = ci;
          }
        }
      }
    }
    __syncthreads();
  }
  if (lane < 16) {
    int row = rowBase + lane;
    #pragma unroll
    for (int q = 0; q < KNN; ++q) idxOut[row * KNN + q] = ik[q];
  }
}

// ---------------------------------------------------------------------------
// EdgeConv: one wave per point. Edge feats gathered straight into A frags,
// two chained f16 WMMAs with LDS staging, max over the 20 valid edges.
// ---------------------------------------------------------------------------
template<int DIN>
__global__ __launch_bounds__(128)
void edge_kernel(const _Float16* __restrict__ xin, int sin,
                 const int* __restrict__ idx,
                 const unsigned int* __restrict__ pwa, const float* __restrict__ ba,
                 const unsigned int* __restrict__ pwb, const float* __restrict__ bb,
                 float* __restrict__ xoutF,
                 _Float16* __restrict__ hcat, int hoff,
                 float* __restrict__ normsOut) {
  constexpr int KCH1 = (2 * DIN + 31) / 32;
  __shared__ _Float16 hbuf[4][32 * 64];
  __shared__ float    obuf[4][32 * 64];
  const int w    = threadIdx.x >> 5;
  const int lane = threadIdx.x & 31;
  const int pt   = blockIdx.x * 4 + w;
  const int p = lane >> 4;
  const int n = lane & 15;
  _Float16* hl = hbuf[w];
  float*    ol = obuf[w];
  const _Float16* xi = xin + (size_t)pt * sin;

  // ---- stage 1: e[32][2*DIN] @ wa -> relu -> h[32][64] in LDS ----
  for (int mt = 0; mt < 2; ++mt) {
    int edge = mt * 16 + n;                       // A-fragment M = lane&15
    int j = (edge < KNN) ? idx[pt * KNN + edge] : pt;
    const _Float16* xj = xin + (size_t)j * sin;
    AFrag a[KCH1];
    #pragma unroll
    for (int ch = 0; ch < KCH1; ++ch) {
      #pragma unroll
      for (int v = 0; v < 8; ++v) {
        int k = ch * 32 + kposf(v, p);
        if (k < DIN) {
          a[ch].h[v] = *(const h2v*)(xi + k);
        } else if (k < 2 * DIN) {
          h2v pj = *(const h2v*)(xj + (k - DIN));
          h2v pi = *(const h2v*)(xi + (k - DIN));
          a[ch].h[v] = pj - pi;
        } else {
          a[ch].u[v] = 0u;
        }
      }
    }
    #pragma unroll
    for (int nt = 0; nt < 4; ++nt) {
      float bias = ba[nt * 16 + n];
      v8f c = {};
      #pragma unroll
      for (int v = 0; v < 8; ++v) c[v] = bias;
      #pragma unroll
      for (int ch = 0; ch < KCH1; ++ch) {
        AFrag bf;
        const u32x4* bp = (const u32x4*)(pwa + ((size_t)(ch * 4 + nt) * 32 + lane) * 8);
        bf.q[0] = bp[0]; bf.q[1] = bp[1];
        c = __builtin_amdgcn_wmma_f32_16x16x32_f16(false, a[ch].v, false, bf.v,
                                                   (short)0, c, false, false);
      }
      #pragma unroll
      for (int v = 0; v < 8; ++v) {
        float hv = fmaxf(c[v], 0.f);
        hl[(mt * 16 + 8 * p + v) * 64 + nt * 16 + n] = (_Float16)hv;
      }
    }
  }
  __syncthreads();

  // ---- stage 2: h[32][64] @ wb + bb -> out[32][64] in LDS ----
  for (int mt = 0; mt < 2; ++mt) {
    AFrag a[2];
    #pragma unroll
    for (int ch = 0; ch < 2; ++ch) {
      const _Float16* hrow = hl + (mt * 16 + n) * 64 + ch * 32 + 8 * p;
      a[ch].q[0] = *(const u32x4*)(hrow);
      a[ch].q[1] = *(const u32x4*)(hrow + 16);
    }
    #pragma unroll
    for (int nt = 0; nt < 4; ++nt) {
      float bias = bb[nt * 16 + n];
      v8f c = {};
      #pragma unroll
      for (int v = 0; v < 8; ++v) c[v] = bias;
      #pragma unroll
      for (int ch = 0; ch < 2; ++ch) {
        AFrag bf;
        const u32x4* bp = (const u32x4*)(pwb + ((size_t)(ch * 4 + nt) * 32 + lane) * 8);
        bf.q[0] = bp[0]; bf.q[1] = bp[1];
        c = __builtin_amdgcn_wmma_f32_16x16x32_f16(false, a[ch].v, false, bf.v,
                                                   (short)0, c, false, false);
      }
      #pragma unroll
      for (int v = 0; v < 8; ++v)
        ol[(mt * 16 + 8 * p + v) * 64 + nt * 16 + n] = c[v];
    }
  }
  __syncthreads();

  // ---- max over the 20 valid edges; lane owns 2 output features ----
  int c0 = lane * 2, c1 = lane * 2 + 1;
  float m0 = -3.4e38f, m1 = -3.4e38f;
  for (int e = 0; e < KNN; ++e) {
    m0 = fmaxf(m0, ol[e * 64 + c0]);
    m1 = fmaxf(m1, ol[e * 64 + c1]);
  }
  if (xoutF) {
    xoutF[(size_t)pt * 64 + c0] = m0;
    xoutF[(size_t)pt * 64 + c1] = m1;
  }
  hcat[(size_t)pt * 192 + hoff + c0] = (_Float16)m0;
  hcat[(size_t)pt * 192 + hoff + c1] = (_Float16)m1;
  if (normsOut) {
    float part = m0 * m0 + m1 * m1;
    __syncthreads();
    ol[lane] = part;
    __syncthreads();
    if (lane == 0) {
      float s = 0.f;
      for (int t = 0; t < 32; ++t) s += ol[t];
      normsOut[pt] = s;
    }
  }
}

// ---------------------------------------------------------------------------
// Generic f16 WMMA GEMM: one wave per (m-tile, group of NTPW n-tiles).
// NTPW is compile-time so the q-loop fully unrolls: no predicated WMMAs,
// no accumulator shuffling. A fragment loaded once per K-chunk, reused
// across NTPW back-to-back WMMAs.
// ---------------------------------------------------------------------------
template<int NTPW>
__global__ __launch_bounds__(256)
void gemm_kernel(const _Float16* __restrict__ A, int lda,
                 const unsigned int* __restrict__ Bp, const float* __restrict__ bias,
                 _Float16* __restrict__ Ch, int ldc, float* __restrict__ Cf,
                 int M, int Kch, int Ntiles, int Nreal, int relu) {
  int w = threadIdx.x >> 5, lane = threadIdx.x & 31;
  int wid = blockIdx.x * 8 + w;
  int ngroups = Ntiles / NTPW;
  int mt = wid / ngroups;
  int ng = wid % ngroups;
  if (mt * 16 >= M) return;
  int p = lane >> 4, n = lane & 15;
  const _Float16* arow = A + (size_t)(mt * 16 + n) * lda;

  v8f acc[NTPW];
  #pragma unroll
  for (int q = 0; q < NTPW; ++q) {
    int col = (ng * NTPW + q) * 16 + n;
    float bv = (col < Nreal) ? bias[col] : 0.f;
    #pragma unroll
    for (int v = 0; v < 8; ++v) acc[q][v] = bv;
  }

  for (int ch = 0; ch < Kch; ++ch) {
    AFrag a;
    const _Float16* ap = arow + ch * 32 + 8 * p;
    a.q[0] = *(const u32x4*)(ap);
    a.q[1] = *(const u32x4*)(ap + 16);
    #pragma unroll
    for (int q = 0; q < NTPW; ++q) {
      int nt = ng * NTPW + q;
      AFrag b;
      const u32x4* bp = (const u32x4*)(Bp + ((size_t)(ch * Ntiles + nt) * 32 + lane) * 8);
      b.q[0] = bp[0]; b.q[1] = bp[1];
      acc[q] = __builtin_amdgcn_wmma_f32_16x16x32_f16(false, a.v, false, b.v,
                                                      (short)0, acc[q], false, false);
    }
  }

  #pragma unroll
  for (int q = 0; q < NTPW; ++q) {
    int col = (ng * NTPW + q) * 16 + n;
    #pragma unroll
    for (int v = 0; v < 8; ++v) {
      int row = mt * 16 + 8 * p + v;
      float val = relu ? fmaxf(acc[q][v], 0.f) : acc[q][v];
      if (Ch) Ch[(size_t)row * ldc + col] = (_Float16)val;
      if (Cf) Cf[(size_t)row * ldc + col] = val;
    }
  }
}

// ---------------------------------------------------------------------------
// log_softmax over 13 logits per row (logits padded to stride 16).
// ---------------------------------------------------------------------------
__global__ __launch_bounds__(256)
void logsoftmax_kernel(const float* __restrict__ logits, float* __restrict__ out) {
  int r = blockIdx.x * blockDim.x + threadIdx.x;
  if (r >= NPTS) return;
  const float* x = logits + (size_t)r * 16;
  float mx = x[0];
  #pragma unroll
  for (int i = 1; i < 13; ++i) mx = fmaxf(mx, x[i]);
  float s = 0.f;
  #pragma unroll
  for (int i = 0; i < 13; ++i) s += __expf(x[i] - mx);
  float lse = mx + __logf(s);
  #pragma unroll
  for (int i = 0; i < 13; ++i) out[(size_t)r * 13 + i] = x[i] - lse;
}

// ---------------------------------------------------------------------------
extern "C" void kernel_launch(void* const* d_in, const int* in_sizes, int n_in,
                              void* d_out, int out_size, void* d_ws, size_t ws_size,
                              hipStream_t stream) {
  const float* feats = (const float*)d_in[0];
  const float* pos   = (const float*)d_in[1];
  // d_in[2] = batch (unused: contiguous equal-size clouds)
  const float* w1a = (const float*)d_in[3];  const float* b1a = (const float*)d_in[4];
  const float* w1b = (const float*)d_in[5];  const float* b1b = (const float*)d_in[6];
  const float* w2a = (const float*)d_in[7];  const float* b2a = (const float*)d_in[8];
  const float* w2b = (const float*)d_in[9];  const float* b2b = (const float*)d_in[10];
  const float* w3a = (const float*)d_in[11]; const float* b3a = (const float*)d_in[12];
  const float* w3b = (const float*)d_in[13]; const float* b3b = (const float*)d_in[14];
  const float* hw1 = (const float*)d_in[15]; const float* hb1 = (const float*)d_in[16];
  const float* hw2 = (const float*)d_in[17]; const float* hb2 = (const float*)d_in[18];
  const float* hw3 = (const float*)d_in[19]; const float* hb3 = (const float*)d_in[20];
  const float* hw4 = (const float*)d_in[21]; const float* hb4 = (const float*)d_in[22];

  char* ws = (char*)d_ws;
  size_t off = 0;
  auto carve = [&](size_t bytes) -> void* {
    void* r = ws + off;
    off = (off + bytes + 255) & ~(size_t)255;
    return r;
  };
  float*        x0f   = (float*)       carve((size_t)NPTS * 8 * 4);
  _Float16*     x0h   = (_Float16*)    carve((size_t)NPTS * 8 * 2);
  float*        norms = (float*)       carve((size_t)NPTS * 4);
  int*          idx   = (int*)         carve((size_t)NPTS * KNN * 4);
  float*        xA    = (float*)       carve((size_t)NPTS * 64 * 4);
  float*        xB    = (float*)       carve((size_t)NPTS * 64 * 4);
  _Float16*     hcat  = (_Float16*)    carve((size_t)NPTS * 192 * 2);
  _Float16*     h1    = (_Float16*)    carve((size_t)NPTS * 1024 * 2);
  _Float16*     h2    = (_Float16*)    carve((size_t)NPTS * 256 * 2);
  _Float16*     h3    = (_Float16*)    carve((size_t)NPTS * 128 * 2);
  float*        lgts  = (float*)       carve((size_t)NPTS * 16 * 4);
  unsigned int* pw1a  = (unsigned int*)carve((size_t)1  * 4  * 256 * 4);
  unsigned int* pw1b  = (unsigned int*)carve((size_t)2  * 4  * 256 * 4);
  unsigned int* pw2a  = (unsigned int*)carve((size_t)4  * 4  * 256 * 4);
  unsigned int* pw2b  = (unsigned int*)carve((size_t)2  * 4  * 256 * 4);
  unsigned int* pw3a  = (unsigned int*)carve((size_t)4  * 4  * 256 * 4);
  unsigned int* pw3b  = (unsigned int*)carve((size_t)2  * 4  * 256 * 4);
  unsigned int* phw1  = (unsigned int*)carve((size_t)6  * 64 * 256 * 4);
  unsigned int* phw2  = (unsigned int*)carve((size_t)32 * 16 * 256 * 4);
  unsigned int* phw3  = (unsigned int*)carve((size_t)8  * 8  * 256 * 4);
  unsigned int* phw4  = (unsigned int*)carve((size_t)4  * 1  * 256 * 4);

  auto pack = [&](const float* W, int Kreal, int Nreal, int Kch, int Nt, unsigned int* outp) {
    int total = Kch * Nt * 256;
    pack_b_kernel<<<(total + 255) / 256, 256, 0, stream>>>(W, Kreal, Nreal, Kch, Nt, outp);
  };
  pack(w1a, 12,   64,   1,  4,  pw1a);
  pack(w1b, 64,   64,   2,  4,  pw1b);
  pack(w2a, 128,  64,   4,  4,  pw2a);
  pack(w2b, 64,   64,   2,  4,  pw2b);
  pack(w3a, 128,  64,   4,  4,  pw3a);
  pack(w3b, 64,   64,   2,  4,  pw3b);
  pack(hw1, 192,  1024, 6,  64, phw1);
  pack(hw2, 1024, 256,  32, 16, phw2);
  pack(hw3, 256,  128,  8,  8,  phw3);
  pack(hw4, 128,  13,   4,  1,  phw4);

  prep_x0_kernel<<<NPTS / 256, 256, 0, stream>>>(feats, pos, x0f, x0h, norms);

  // Layer 1
  knn_kernel<8><<<256, 128, 0, stream>>>(x0f, norms, idx);
  edge_kernel<6><<<NPTS / 4, 128, 0, stream>>>(x0h, 8, idx, pw1a, b1a, pw1b, b1b,
                                               xA, hcat, 0, norms);
  // Layer 2
  knn_kernel<64><<<256, 128, 0, stream>>>(xA, norms, idx);
  edge_kernel<64><<<NPTS / 4, 128, 0, stream>>>(hcat, 192, idx, pw2a, b2a, pw2b, b2b,
                                                xB, hcat, 64, norms);
  // Layer 3
  knn_kernel<64><<<256, 128, 0, stream>>>(xB, norms, idx);
  edge_kernel<64><<<NPTS / 4, 128, 0, stream>>>(hcat + 64, 192, idx, pw3a, b3a, pw3b, b3b,
                                                nullptr, hcat, 128, nullptr);

  // Head MLP: waves = (M/16) * (Ntiles/NTPW), 8 waves per block.
  gemm_kernel<4><<<(NPTS / 16) * (64 / 4) / 8, 256, 0, stream>>>(hcat, 192, phw1, hb1,
      h1, 1024, nullptr, NPTS, 6, 64, 1024, 1);
  gemm_kernel<4><<<(NPTS / 16) * (16 / 4) / 8, 256, 0, stream>>>(h1, 1024, phw2, hb2,
      h2, 256, nullptr, NPTS, 32, 16, 256, 1);
  gemm_kernel<4><<<(NPTS / 16) * (8 / 4) / 8, 256, 0, stream>>>(h2, 256, phw3, hb3,
      h3, 128, nullptr, NPTS, 8, 8, 128, 1);
  gemm_kernel<1><<<(NPTS / 16) * 1 / 8, 256, 0, stream>>>(h3, 128, phw4, hb4,
      nullptr, 16, lgts, NPTS, 4, 1, 13, 0);

  logsoftmax_kernel<<<NPTS / 256, 256, 0, stream>>>(lgts, (float*)d_out);
}
